// ReLU_Feat_Posenc_1597727834798
// MI455X (gfx1250) — matrix operand order repaired and checked by reading, not compiled, not run on hardware
//
#include <hip/hip_runtime.h>

// ---------------------------------------------------------------------------
// Fused posenc + bilinear feature gather + 5-layer MLP for gfx1250 (MI455X)
// - MLP on v_wmma_f32_16x16x32_f16 (f16 in, f32 accumulate)
// - Weights staged once/block to LDS as f16, transposed [N][K]
// - M=32 per wave (two A fragments share each B fragment -> 2x B reuse)
// - B fragments double-buffered so LDS latency overlaps WMMA execution
// - Grid-stride tile loop amortizes weight staging over 4 tiles per wave
// ---------------------------------------------------------------------------

typedef _Float16 v16h __attribute__((ext_vector_type(16)));
typedef _Float16 v8h  __attribute__((ext_vector_type(8)));
typedef float    v8f  __attribute__((ext_vector_type(8)));

union ABfrag { v16h v; v8h h[2]; };

// ---- geometry -------------------------------------------------------------
#define GRID_N   512
#define FEAT     100
#define PE_DIM   66          // 2*(2*16+1)
#define FFIN     166
#define K0PAD    192         // FFIN padded to multiple of 32
#define HID      128
#define NPTS     1048576
#define WAVES    8
#define TPB      256
#define MTILE    32          // points per wave per tile
#define NBLOCKS  1024        // 1024*8 wave-slots, 4 tiles each

// ---- LDS layout (units: halves unless noted) ------------------------------
#define STRA     200         // activation row stride (192 + 8 pad) -> conflict-free b128
#define STRW0    200         // layer-0 weight K-stride (192 + 8 pad)
#define STRW     136         // layers 1..4 weight K-stride (128 + 8 pad)
#define WT0_OFF  0
#define WT0_SZ   (HID * STRW0)                 // 25600
#define WT1_OFF  (WT0_OFF + WT0_SZ)
#define WT_SZ    (HID * STRW)                  // 17408
#define WT4_OFF  (WT1_OFF + 3 * WT_SZ)         // 77824
#define WT4_SZ   (16 * STRW)                   // 2176
#define WT_END   (WT4_OFF + WT4_SZ)            // 80000 halves
#define BIAS_OFF_B   (WT_END * 2)              // 160000 bytes
#define BIAS_FLOATS  528                       // 4*128 + 16
#define ACT_OFF_B    (BIAS_OFF_B + BIAS_FLOATS * 4)   // 162112 bytes
#define ACT_PER_WAVE (MTILE * STRA)            // 6400 halves / wave
#define SMEM_BYTES   (ACT_OFF_B + WAVES * ACT_PER_WAVE * 2)  // 264512 B < 320KB/WGP

// ---------------------------------------------------------------------------
// One MLP layer on a 32-point tile owned by a single wave32.
//   actIn : LDS, row-major [32 x K], row stride STRA halves
//   wt    : LDS, transposed weights [Nout x K], row stride strideW halves
//   A frag (16x32 f16, ISA 7.12.2): lane L -> row L&15, kbase = k0 + (L>=16)*8,
//           v0..3 = K[kbase..kbase+8), v4..7 = K[kbase+16..kbase+24)
//   B frag (32x16 f16): lane L -> col L&15, same K order as A frag
//   C/D:    lane L -> col L&15, VGPR j -> row j + (L>=16)*8
// ---------------------------------------------------------------------------
template<int KSTEPS, int NTILES, bool LAST>
__device__ __forceinline__ void mlp_layer(const _Float16* actIn,
                                          const _Float16* wt, int strideW,
                                          const float* bias,
                                          _Float16* actOut,
                                          float* gout, int basePoint, int lane)
{
    const int row = lane & 15;
    const int hi  = lane >> 4;      // 0 or 1
    const _Float16* wrow = wt + row * strideW + hi * 8;

    // Load all A fragments (both 16-row halves of the 32-point tile) into
    // registers first, so the activation tile can be overwritten in place.
    ABfrag a[2][KSTEPS];
#pragma unroll
    for (int r = 0; r < 2; ++r) {
#pragma unroll
        for (int ks = 0; ks < KSTEPS; ++ks) {
            const _Float16* p = actIn + (r * 16 + row) * STRA + ks * 32 + hi * 8;
            a[r][ks].h[0] = *(const v8h*)(p);
            a[r][ks].h[1] = *(const v8h*)(p + 16);
        }
    }

#pragma unroll
    for (int nt = 0; nt < NTILES; ++nt) {
        v8f acc0 = {0.f, 0.f, 0.f, 0.f, 0.f, 0.f, 0.f, 0.f};
        v8f acc1 = {0.f, 0.f, 0.f, 0.f, 0.f, 0.f, 0.f, 0.f};

        // Double-buffered B fragments: keep the next k-step's LDS load in
        // flight while the current WMMA pair executes (avoids dscnt==0 drains).
        ABfrag b[2];
        {
            const _Float16* p = wrow + nt * 16 * strideW;
            b[0].h[0] = *(const v8h*)(p);
            b[0].h[1] = *(const v8h*)(p + 16);
        }
#pragma unroll
        for (int ks = 0; ks < KSTEPS; ++ks) {
            if (ks + 1 < KSTEPS) {
                const _Float16* p = wrow + nt * 16 * strideW + (ks + 1) * 32;
                b[(ks + 1) & 1].h[0] = *(const v8h*)(p);
                b[(ks + 1) & 1].h[1] = *(const v8h*)(p + 16);
            }
            acc0 = __builtin_amdgcn_wmma_f32_16x16x32_f16(
                false, a[0][ks].v, false, b[ks & 1].v, (short)0, acc0, false, false);
            acc1 = __builtin_amdgcn_wmma_f32_16x16x32_f16(
                false, a[1][ks].v, false, b[ks & 1].v, (short)0, acc1, false, false);
        }

        const float bn = bias[nt * 16 + row];
        if constexpr (!LAST) {
#pragma unroll
            for (int j = 0; j < 8; ++j) {
                float x0 = acc0[j] + bn;
                float x1 = acc1[j] + bn;
                x0 = fmaxf(x0, 0.01f * x0);     // leaky relu (slope>0 => max form)
                x1 = fmaxf(x1, 0.01f * x1);
                actOut[(     j + hi * 8) * STRA + nt * 16 + row] = (_Float16)x0;
                actOut[(16 + j + hi * 8) * STRA + nt * 16 + row] = (_Float16)x1;
            }
        } else {
            if (row < 3) {
#pragma unroll
                for (int j = 0; j < 8; ++j) {
                    gout[(size_t)(basePoint +      j + hi * 8) * 3 + row] = acc0[j] + bn;
                    gout[(size_t)(basePoint + 16 + j + hi * 8) * 3 + row] = acc1[j] + bn;
                }
            }
        }
    }
}

__global__ __launch_bounds__(TPB) void nerf_mlp_kernel(
    const float* __restrict__ UV, const float* __restrict__ feat,
    const float* __restrict__ W0, const float* __restrict__ b0,
    const float* __restrict__ W1, const float* __restrict__ b1,
    const float* __restrict__ W2, const float* __restrict__ b2,
    const float* __restrict__ W3, const float* __restrict__ b3,
    const float* __restrict__ W4, const float* __restrict__ b4,
    float* __restrict__ out)
{
    extern __shared__ char smem[];
    _Float16* sh   = (_Float16*)smem;
    _Float16* wt0  = sh + WT0_OFF;
    _Float16* wt1  = sh + WT1_OFF;
    _Float16* wt2  = wt1 + WT_SZ;
    _Float16* wt3  = wt2 + WT_SZ;
    _Float16* wt4  = sh + WT4_OFF;
    float* biasLds = (float*)(smem + BIAS_OFF_B);
    _Float16* actAll = (_Float16*)(smem + ACT_OFF_B);

    const int tid = threadIdx.x;

    // ---- cooperative weight staging: fp32 -> f16, transposed to [N][K] ----
    for (int idx = tid; idx < HID * K0PAD; idx += TPB) {
        int n = idx / K0PAD, k = idx - n * K0PAD;
        wt0[n * STRW0 + k] = (k < FFIN) ? (_Float16)W0[k * HID + n] : (_Float16)0.f;
    }
    for (int idx = tid; idx < HID * HID; idx += TPB) {
        int n = idx >> 7, k = idx & 127;
        wt1[n * STRW + k] = (_Float16)W1[k * HID + n];
    }
    for (int idx = tid; idx < HID * HID; idx += TPB) {
        int n = idx >> 7, k = idx & 127;
        wt2[n * STRW + k] = (_Float16)W2[k * HID + n];
    }
    for (int idx = tid; idx < HID * HID; idx += TPB) {
        int n = idx >> 7, k = idx & 127;
        wt3[n * STRW + k] = (_Float16)W3[k * HID + n];
    }
    for (int idx = tid; idx < 16 * HID; idx += TPB) {
        int n = idx >> 7, k = idx & 127;
        wt4[n * STRW + k] = (n < 3) ? (_Float16)W4[k * 3 + n] : (_Float16)0.f;
    }
    for (int i = tid; i < 512; i += TPB)
        biasLds[i] = (i < 128) ? b0[i] : (i < 256) ? b1[i - 128]
                   : (i < 384) ? b2[i - 256] : b3[i - 384];
    if (tid < 16) biasLds[512 + tid] = (tid < 3) ? b4[tid] : 0.f;

    __syncthreads();   // weights visible to all waves; act tiles are wave-private

    const int wave = tid >> 5, lane = tid & 31;
    _Float16* act = actAll + wave * ACT_PER_WAVE;
    const int tilesTotal = NPTS / MTILE;            // 32768
    const int waveSlots  = gridDim.x * WAVES;       // 8192 -> 4 tiles per wave

#pragma unroll 1
    for (int tile = blockIdx.x * WAVES + wave; tile < tilesTotal; tile += waveSlots) {
        const int basePoint = tile * MTILE;

        // ---- input staging: 1 lane per point, 32 points per wave ----
        {
            const int gp = basePoint + lane;
            const float u = UV[2 * gp + 0];
            const float v = UV[2 * gp + 1];
            _Float16* arow = act + lane * STRA;

            // posenc: [u, v, sin(ku),sin(kv),cos(ku),cos(kv)] for k=1..2^15
            arow[0] = (_Float16)u;
            arow[1] = (_Float16)v;
            for (int f = 0; f < 16; ++f) {
                float k = (float)(1 << f);
                arow[2 + 4 * f + 0] = (_Float16)__sinf(k * u);
                arow[2 + 4 * f + 1] = (_Float16)__sinf(k * v);
                arow[2 + 4 * f + 2] = (_Float16)__cosf(k * u);
                arow[2 + 4 * f + 3] = (_Float16)__cosf(k * v);
            }

            // bilinear gather of 100-dim features (grid L2-resident: 105MB<192MB)
            const float iu = (float)(GRID_N - 1) * u;
            const float iv = (float)(GRID_N - 1) * v;
            const float fu0 = floorf(iu), fv0 = floorf(iv);
            const int iU0 = (int)fu0,        iV0 = (int)fv0;
            const int iU1 = (int)ceilf(iu),  iV1 = (int)ceilf(iv);
            const float fu = iu - fu0, fv = iv - fv0;
            const float w00 = (1.f - fu) * (1.f - fv);
            const float w10 = fu * (1.f - fv);
            const float w01 = (1.f - fu) * fv;
            const float w11 = fu * fv;
            const float* f00 = feat + ((size_t)iU0 * GRID_N + iV0) * FEAT;
            const float* f10 = feat + ((size_t)iU1 * GRID_N + iV0) * FEAT;
            const float* f01 = feat + ((size_t)iU0 * GRID_N + iV1) * FEAT;
            const float* f11 = feat + ((size_t)iU1 * GRID_N + iV1) * FEAT;
            for (int d = 0; d < FEAT; d += 2) {
                float2 a0 = *(const float2*)(f00 + d);
                float2 a1 = *(const float2*)(f10 + d);
                float2 a2 = *(const float2*)(f01 + d);
                float2 a3 = *(const float2*)(f11 + d);
                arow[PE_DIM + d]     = (_Float16)(a0.x * w00 + a1.x * w10 + a2.x * w01 + a3.x * w11);
                arow[PE_DIM + d + 1] = (_Float16)(a0.y * w00 + a1.y * w10 + a2.y * w01 + a3.y * w11);
            }
            for (int j = FFIN; j < K0PAD; ++j) arow[j] = (_Float16)0.f;
        }

        // ---- 5 layers, all on WMMA; activation tile reused in place ----
        // (wave-private LDS + in-order DS per wave => no barriers needed)
        mlp_layer<6, 8, false>(act, wt0, STRW0, biasLds +   0, act, nullptr, 0, lane);
        mlp_layer<4, 8, false>(act, wt1, STRW,  biasLds + 128, act, nullptr, 0, lane);
        mlp_layer<4, 8, false>(act, wt2, STRW,  biasLds + 256, act, nullptr, 0, lane);
        mlp_layer<4, 8, false>(act, wt3, STRW,  biasLds + 384, act, nullptr, 0, lane);
        mlp_layer<4, 1, true >(act, wt4, STRW,  biasLds + 512, nullptr, out, basePoint, lane);
    }
}

extern "C" void kernel_launch(void* const* d_in, const int* in_sizes, int n_in,
                              void* d_out, int out_size, void* d_ws, size_t ws_size,
                              hipStream_t stream) {
    (void)in_sizes; (void)n_in; (void)d_ws; (void)ws_size; (void)out_size;
    const float* UV   = (const float*)d_in[0];
    const float* feat = (const float*)d_in[1];
    const float* W0   = (const float*)d_in[2];
    const float* b0   = (const float*)d_in[3];
    const float* W1   = (const float*)d_in[4];
    const float* b1   = (const float*)d_in[5];
    const float* W2   = (const float*)d_in[6];
    const float* b2   = (const float*)d_in[7];
    const float* W3   = (const float*)d_in[8];
    const float* b3   = (const float*)d_in[9];
    const float* W4   = (const float*)d_in[10];
    const float* b4   = (const float*)d_in[11];
    float* out = (float*)d_out;

    // Opt in to >default dynamic LDS (258 KB; WGP supports 320 KB).
    (void)hipFuncSetAttribute(reinterpret_cast<const void*>(nerf_mlp_kernel),
                              hipFuncAttributeMaxDynamicSharedMemorySize,
                              SMEM_BYTES);

    nerf_mlp_kernel<<<NBLOCKS, TPB, SMEM_BYTES, stream>>>(
        UV, feat, W0, b0, W1, b1, W2, b2, W3, b3, W4, b4, out);
}
